// SimpleRingAttention_76312978915795
// MI455X (gfx1250) — matrix operand order repaired and checked by reading, not compiled, not run on hardware
//
#include <hip/hip_runtime.h>

// ---- CDNA5 WMMA types -------------------------------------------------------
typedef __attribute__((ext_vector_type(16))) _Float16 v16h;
typedef __attribute__((ext_vector_type(8)))  _Float16 v8h;
typedef __attribute__((ext_vector_type(8)))  float    v8f;
typedef __attribute__((ext_vector_type(4)))  int      v4i;

#if __has_builtin(__builtin_amdgcn_exp2f)
#define EXP2F __builtin_amdgcn_exp2f
#else
#define EXP2F exp2f
#endif

static __device__ __forceinline__ v8f wmma_f16(v16h a, v16h b, v8f c) {
  return __builtin_amdgcn_wmma_f32_16x16x32_f16(false, a, false, b, (short)0, c,
                                                false, false);
}

// DPP xor-shuffle within 16-lane rows (ROW_XMASK = 0x160 | mask): pure VALU.
template <int CTRL>
static __device__ __forceinline__ float dppmovf(float x) {
  int i = __builtin_bit_cast(int, x);
  int r = __builtin_amdgcn_update_dpp(i, i, CTRL, 0xF, 0xF, true);
  return __builtin_bit_cast(float, r);
}
static __device__ __forceinline__ float rowmax16(float x) {
  x = fmaxf(x, dppmovf<0x161>(x));   // xor 1
  x = fmaxf(x, dppmovf<0x162>(x));   // xor 2
  x = fmaxf(x, dppmovf<0x164>(x));   // xor 4
  x = fmaxf(x, dppmovf<0x168>(x));   // xor 8
  return x;
}

static __device__ __forceinline__ void put4(v16h& dst, int i, float4 f, float s) {
  dst[i + 0] = (_Float16)(f.x * s);
  dst[i + 1] = (_Float16)(f.y * s);
  dst[i + 2] = (_Float16)(f.z * s);
  dst[i + 3] = (_Float16)(f.w * s);
}

// ---- Async global->LDS (ASYNCcnt path) --------------------------------------
#if __has_builtin(__builtin_amdgcn_global_load_async_to_lds_b128)
#define HAVE_ASYNC_LDS 1
typedef __attribute__((address_space(1))) v4i* as1_v4i;
typedef __attribute__((address_space(3))) v4i* as3_v4i;
#endif

static __device__ __forceinline__ void copy16_g2l(const void* g, void* l) {
#ifdef HAVE_ASYNC_LDS
  __builtin_amdgcn_global_load_async_to_lds_b128((as1_v4i)g, (as3_v4i)l, 0, 0);
#else
  *(int4*)l = *(const int4*)g;
#endif
}
static __device__ __forceinline__ void wait_async_lds() {
#ifdef HAVE_ASYNC_LDS
#if __has_builtin(__builtin_amdgcn_s_wait_asynccnt)
  __builtin_amdgcn_s_wait_asynccnt(0);
#else
  asm volatile("s_wait_asynccnt 0" ::: "memory");
#endif
#endif
}

// ---- Problem constants ------------------------------------------------------
static constexpr int Bsz = 2;
static constexpr int Sq  = 4096;
static constexpr int Hh  = 8;
static constexpr int Dd  = 64;
static constexpr int ROWSTRIDE = Hh * Dd;    // 512 floats between rows (s)
static constexpr int KB  = 64;               // keys per inner block
static constexpr int QT  = 64;               // query rows per workgroup
static constexpr int NW  = 4;                // waves
static constexpr int NIT = Sq / KB;          // 64 key blocks
static constexpr int TOT4 = Bsz * Sq * Hh * Dd / 4;  // float4 count per tensor

// ---- Pre-pass: K -> f16 [bh][s][d], V -> f16 transposed [bh][d][s] ----------
__global__ void cvt_f16_kernel(const float* __restrict__ Kg,
                               const float* __restrict__ Vg,
                               _Float16* __restrict__ Kh,
                               _Float16* __restrict__ Vth) {
  int i = blockIdx.x * blockDim.x + threadIdx.x;
  if (i >= TOT4) return;
  const int d4 = i & 15;          // d/4
  const int h  = (i >> 4) & 7;
  const int s  = (i >> 7) & 4095;
  const int b  = i >> 19;
  float4 kf = ((const float4*)Kg)[i];
  float4 vf = ((const float4*)Vg)[i];
  const size_t kdst = ((size_t)(b * Hh + h) * Sq + s) * Dd + d4 * 4;
  Kh[kdst + 0] = (_Float16)kf.x;
  Kh[kdst + 1] = (_Float16)kf.y;
  Kh[kdst + 2] = (_Float16)kf.z;
  Kh[kdst + 3] = (_Float16)kf.w;
  const size_t vbase = ((size_t)(b * Hh + h) * Dd + d4 * 4) * Sq + s;
  Vth[vbase + 0 * Sq] = (_Float16)vf.x;
  Vth[vbase + 1 * Sq] = (_Float16)vf.y;
  Vth[vbase + 2 * Sq] = (_Float16)vf.z;
  Vth[vbase + 3 * Sq] = (_Float16)vf.w;
}

// ---- Flash attention main kernel -------------------------------------------
template <bool USE_WS>
__launch_bounds__(NW * 32)
__global__ void ring_attn_fa_kernel(const float* __restrict__ Qg,
                                    const float* __restrict__ Kg,
                                    const float* __restrict__ Vg,
                                    const _Float16* __restrict__ Khg,
                                    const _Float16* __restrict__ Vthg,
                                    float* __restrict__ Og) {
  __shared__ __align__(32) _Float16 sK[2][KB * Dd];    // [key][d]   2 x 8 KB
  __shared__ __align__(32) _Float16 sVt[2][Dd * KB];   // [d][key]   2 x 8 KB
  __shared__ __align__(32) _Float16 sP[NW][16 * KB];   // per-wave P 4 x 2 KB

  const int tid   = threadIdx.x;
  const int lane  = tid & 31;
  const int w     = tid >> 5;
  const int n16   = lane & 15;
  const int halfw = lane >> 4;

  const int tiles = Sq / QT;
  const int tile  = blockIdx.x % tiles;
  const int bh    = blockIdx.x / tiles;
  const int h     = bh % Hh;
  const int b     = bh / Hh;
  const size_t bh_off = (size_t)b * Sq * ROWSTRIDE + (size_t)h * Dd;

  // ---- Q fragments (A layout), pre-scaled by 1/sqrt(D) * log2(e) -----------
  const float QSCALE = 0.125f * 1.44269504088896340736f;
  const int   q0     = tile * QT + w * 16;
  const float* qrow  = Qg + bh_off + (size_t)(q0 + n16) * ROWSTRIDE;
  v16h aq[2];
#pragma unroll
  for (int f = 0; f < 2; ++f) {
    const float* pa = qrow + 32 * f + 16 * halfw;
    const float* pb = qrow + 32 * f + 8 + 16 * halfw;
    put4(aq[f], 0,  *(const float4*)(pa),     QSCALE);
    put4(aq[f], 4,  *(const float4*)(pa + 4), QSCALE);
    put4(aq[f], 8,  *(const float4*)(pb),     QSCALE);
    put4(aq[f], 12, *(const float4*)(pb + 4), QSCALE);
  }

  // ---- Flash state ----------------------------------------------------------
  const v8f vzero = {0.f, 0.f, 0.f, 0.f, 0.f, 0.f, 0.f, 0.f};
  v8f  o[4] = {vzero, vzero, vzero, vzero};
  v8f  lacc = vzero;                       // row sums via ones-WMMA
  float mrow[8];
#pragma unroll
  for (int r = 0; r < 8; ++r) mrow[r] = -1e30f;

  v16h vone;
#pragma unroll
  for (int i = 0; i < 16; ++i) vone[i] = (_Float16)1.0f;

  // ---- Staging --------------------------------------------------------------
  auto stage = [&](int it, int buf) {
    const int kb = it * KB;
    _Float16* dK = &sK[buf][0];
    _Float16* dV = &sVt[buf][0];
    if constexpr (USE_WS) {
      // K block: 64 keys x 64 halves, fully contiguous (8 KB) in Khg.
      const char* gK = (const char*)(Khg + ((size_t)bh * Sq + kb) * Dd);
#pragma unroll
      for (int i = 0; i < 4; ++i)
        copy16_g2l(gK + tid * 64 + i * 16, (char*)dK + tid * 64 + i * 16);
      // V block: 64 d-rows x 64 keys, rows strided by Sq halves in Vthg.
      const int r  = tid >> 1;
      const int c0 = (tid & 1) * 4;
      const char* gV = (const char*)(Vthg + ((size_t)bh * Dd + r) * Sq + kb);
#pragma unroll
      for (int i = 0; i < 4; ++i)
        copy16_g2l(gV + (c0 + i) * 16, (char*)(dV + r * KB) + (c0 + i) * 16);
    } else {
      // Self-contained path: stage from f32 globals with in-kernel convert.
      const int key = tid >> 1;
      const int d0  = (tid & 1) * 32;
      const float* krow = Kg + bh_off + (size_t)(kb + key) * ROWSTRIDE + d0;
      const float* vrow = Vg + bh_off + (size_t)(kb + key) * ROWSTRIDE + d0;
      v16h h0, h1;
#pragma unroll
      for (int i = 0; i < 4; ++i) put4(h0, 4 * i, *(const float4*)(krow + 4 * i), 1.0f);
#pragma unroll
      for (int i = 0; i < 4; ++i) put4(h1, 4 * i, *(const float4*)(krow + 16 + 4 * i), 1.0f);
      *(v16h*)&dK[key * Dd + d0]      = h0;
      *(v16h*)&dK[key * Dd + d0 + 16] = h1;
#pragma unroll
      for (int i = 0; i < 8; ++i) {
        float4 vf = *(const float4*)(vrow + 4 * i);
        dV[(d0 + 4 * i + 0) * KB + key] = (_Float16)vf.x;
        dV[(d0 + 4 * i + 1) * KB + key] = (_Float16)vf.y;
        dV[(d0 + 4 * i + 2) * KB + key] = (_Float16)vf.z;
        dV[(d0 + 4 * i + 3) * KB + key] = (_Float16)vf.w;
      }
    }
  };

  stage(0, 0);
  wait_async_lds();
  __syncthreads();

  int buf = 0;
  for (int it = 0; it < NIT; ++it) {
    if (it + 1 < NIT) stage(it + 1, buf ^ 1);   // overlap with compute below

    const _Float16* cK = &sK[buf][0];
    const _Float16* cV = &sVt[buf][0];

    // ---- Scores: four 16x16 tiles over 64 keys ----------------------------
    v8f sc[4];
#pragma unroll
    for (int t = 0; t < 4; ++t) {
      const int keyl = t * 16 + n16;
      v16h bk0 = *(const v16h*)&cK[keyl * Dd + 16 * halfw];        // d 0..31
      v16h bk1 = *(const v16h*)&cK[keyl * Dd + 32 + 16 * halfw];   // d 32..63
      v8f c = vzero;
      c = wmma_f16(aq[0], bk0, c);
      c = wmma_f16(aq[1], bk1, c);
      sc[t] = c;
    }

    // ---- Online softmax: row max via DPP, row sum via ones-WMMA -----------
    float mnew[8], alpha[8];
#pragma unroll
    for (int r = 0; r < 8; ++r) {
      float t01 = fmaxf(sc[0][r], sc[1][r]);
      float t23 = fmaxf(sc[2][r], sc[3][r]);
      float t   = rowmax16(fmaxf(t01, t23));
      mnew[r]   = fmaxf(mrow[r], t);
      alpha[r]  = EXP2F(mrow[r] - mnew[r]);
      mrow[r]   = mnew[r];
    }

    _Float16* pw = &sP[w][0];
#pragma unroll
    for (int r = 0; r < 8; ++r) {
      const int row = r + 8 * halfw;             // C layout: M = r + 8*half
#pragma unroll
      for (int t = 0; t < 4; ++t) {
        float p = EXP2F(sc[t][r] - mnew[r]);
        pw[row * KB + t * 16 + n16] = (_Float16)p;
      }
#pragma unroll
      for (int j = 0; j < 4; ++j) o[j][r] *= alpha[r];
      lacc[r] *= alpha[r];
    }
    asm volatile("s_wait_dscnt 0" ::: "memory");  // in-wave LDS RAW fence

    // ---- P A-frags (keys 0..31 and 32..63) --------------------------------
    v8h p0l = *(const v8h*)&pw[n16 * KB + 16 * halfw];
    v8h p0h = *(const v8h*)&pw[n16 * KB + 8 + 16 * halfw];
    v8h p1l = *(const v8h*)&pw[n16 * KB + 32 + 16 * halfw];
    v8h p1h = *(const v8h*)&pw[n16 * KB + 40 + 16 * halfw];
    v16h ap0 = __builtin_shufflevector(p0l, p0h, 0, 1, 2, 3, 4, 5, 6, 7,
                                       8, 9, 10, 11, 12, 13, 14, 15);
    v16h ap1 = __builtin_shufflevector(p1l, p1h, 0, 1, 2, 3, 4, 5, 6, 7,
                                       8, 9, 10, 11, 12, 13, 14, 15);

    // ---- O += P*V; l += P*1 ----------------------------------------------
#pragma unroll
    for (int j = 0; j < 4; ++j) {
      const int drow = 16 * j + n16;
      v16h bv0 = *(const v16h*)&cV[drow * KB + 16 * halfw];        // keys 0..31
      v16h bv1 = *(const v16h*)&cV[drow * KB + 32 + 16 * halfw];   // keys 32..63
      o[j] = wmma_f16(ap0, bv0, o[j]);
      o[j] = wmma_f16(ap1, bv1, o[j]);
    }
    lacc = wmma_f16(ap0, vone, lacc);
    lacc = wmma_f16(ap1, vone, lacc);

    wait_async_lds();
    __syncthreads();
    buf ^= 1;
  }

  // ---- Epilogue -------------------------------------------------------------
#pragma unroll
  for (int r = 0; r < 8; ++r) {
    const float inv = 1.0f / lacc[r];
    const int   row = q0 + r + 8 * halfw;
    float* op = Og + bh_off + (size_t)row * ROWSTRIDE;
#pragma unroll
    for (int j = 0; j < 4; ++j) op[16 * j + n16] = o[j][r] * inv;
  }
}

// ---- Host launch ------------------------------------------------------------
extern "C" void kernel_launch(void* const* d_in, const int* in_sizes, int n_in,
                              void* d_out, int out_size, void* d_ws, size_t ws_size,
                              hipStream_t stream) {
  (void)in_sizes; (void)n_in; (void)out_size;
  const float* q = (const float*)d_in[0];
  const float* k = (const float*)d_in[1];
  const float* v = (const float*)d_in[2];
  float* out = (float*)d_out;

  const int blocks = Bsz * Hh * (Sq / QT);             // 1024 workgroups
  const size_t halfelems = (size_t)Bsz * Hh * Sq * Dd; // 4 Mi halves per tensor
  const size_t need = 2 * halfelems * sizeof(unsigned short);

  if (ws_size >= need) {
    _Float16* Kh  = (_Float16*)d_ws;
    _Float16* Vth = Kh + halfelems;
    cvt_f16_kernel<<<TOT4 / 256, 256, 0, stream>>>(k, v, Kh, Vth);
    ring_attn_fa_kernel<true><<<blocks, NW * 32, 0, stream>>>(q, k, v, Kh, Vth, out);
  } else {
    ring_attn_fa_kernel<false><<<blocks, NW * 32, 0, stream>>>(q, k, v, nullptr,
                                                               nullptr, out);
  }
}
